// CausalSelfAttention_42056319762674
// MI455X (gfx1250) — compile-verified, matrix operand order checked by reading, not run on hardware
//
#include <hip/hip_runtime.h>
#include <hip/hip_bf16.h>

typedef __bf16 bf16;
typedef __attribute__((ext_vector_type(16))) __bf16 bf16x16;
typedef __attribute__((ext_vector_type(8)))  float  f32x8;
typedef __attribute__((ext_vector_type(4)))  unsigned su32x4;   // D# group0 (SGPR quad)
typedef __attribute__((ext_vector_type(8)))  unsigned su32x8;   // D# group1 (SGPR octet)

#define WMMA_BF16(A, B, C) \
  __builtin_amdgcn_wmma_f32_16x16x32_bf16(false, (A), false, (B), (short)0, (C), false, false)

// ---------------------------------------------------------------------------
// Fragment loader: 16-bit A/B layout (ISA 7.12.2). For a lane, elements
// j=0..7  come from rowBase[kOff + j]       (K = kBase + 0..7)
// j=8..15 come from rowBase[kOff + 8 + j]  (K = kBase + 16..23)
// where kOff = (lane>>4)*8. Rows are 16B aligned -> two ds_load_b128.
// ---------------------------------------------------------------------------
__device__ __forceinline__ bf16x16 load_frag(const bf16* rowBase, int kOff) {
  union { bf16x16 v; uint4 q[2]; } u;
  u.q[0] = *(const uint4*)(rowBase + kOff);
  u.q[1] = *(const uint4*)(rowBase + kOff + 16);
  return u.v;
}

__device__ __forceinline__ float ldIn(const float* p) { return *p; }
__device__ __forceinline__ float ldIn(const bf16*  p) { return (float)*p; }
__device__ __forceinline__ void  stOut(float* p, float v) { *p = v; }
__device__ __forceinline__ void  stOut(bf16*  p, float v) { *p = (bf16)v; }

// ---------------------------------------------------------------------------
// TDM: issue a 2D bf16 tile load (64 rows x 64 elems, global row stride in
// elements) into LDS with padded rows (pad 4 DWORDs after every 32 DWORDs ->
// 144B LDS row stride == the 72-half padded rows used by the compute code).
// Descriptor bitfields per CDNA5 ISA §8.3/8.4 (D# group 0/1); groups 2/3 are
// NULL (2D tensor). Issued via inline asm (portable across toolchain builtin
// arity split); completion tracked with TENSORcnt.
// ---------------------------------------------------------------------------
__device__ __forceinline__ void tdm_load_tile_64x64_bf16(unsigned ldsByteAddr,
                                                         const bf16* gsrc,
                                                         unsigned rowStrideElems) {
  const unsigned long long ga = (unsigned long long)(uintptr_t)gsrc;
  su32x4 g0;
  g0.x = 1u;                                              // count=1 (valid), user mode
  g0.y = ldsByteAddr;                                     // lds_addr[31:0]
  g0.z = (unsigned)(ga & 0xFFFFFFFFull);                  // global_addr[31:0]
  g0.w = (unsigned)((ga >> 32) & 0x01FFFFFFull)           // global_addr[56:32]
       | (2u << 30);                                      // type=2 ("image")
  su32x8 g1;
  g1[0] = (1u << 16)      // data_size = 1 -> 2 bytes
        | (1u << 20)      // pad_enable
        | (4u << 22)      // pad_interval: 4 -> every 32 DWORDs (128B = one row)
        | (3u << 25);     // pad_amount:  3 -> 4 DWORDs (16B)  => 144B row pitch
  g1[1] = 64u << 16;      // tensor_dim0 = 64  (bits 79:48, low half)
  g1[2] = 64u << 16;      // tensor_dim1 = 64  (bits 111:80, low half)
  g1[3] = 64u << 16;      // tile_dim0   = 64  (bits 127:112)
  g1[4] = 64u;            // tile_dim1   = 64  (bits 143:128), tile_dim2 = 0
  g1[5] = rowStrideElems; // tensor_dim0_stride[31:0] (bits 207:160)
  g1[6] = 0u;             // stride high bits / tensor_dim1_stride = 0
  g1[7] = 0u;
  asm volatile("tensor_load_to_lds %0, %1" :: "s"(g0), "s"(g1) : "memory");
}

// ---------------------------------------------------------------------------
// Generic GEMM:  C[M,N] = A[M,K] @ B[K,N] + bias[N]
// A row-major (fp32 or bf16), B row-major fp32, C fp32 or bf16.
// Block tile 128x128, 256 threads = 8 waves (4 in M x 2 in N),
// each wave computes 32x64 via v_wmma_f32_16x16x32_bf16, K-step 32.
// ---------------------------------------------------------------------------
template <typename InT, typename OutT>
__global__ __launch_bounds__(256)
void gemm_bf16_wmma(const InT* __restrict__ A, const float* __restrict__ B,
                    const float* __restrict__ bias, OutT* __restrict__ C,
                    int K, int N) {
  __shared__ __align__(16) bf16 As[128 * 40];      // As[m][k], 80B rows
  __shared__ __align__(16) bf16 Bs[128 * 40];      // transposed: Bs[n][k]

  const int tid   = threadIdx.x;
  const int lane  = tid & 31;
  const int wave  = tid >> 5;
  const int waveM = wave & 3;
  const int waveN = wave >> 2;
  const int l16   = lane & 15;
  const int lhalf = lane >> 4;
  const int kOff  = lhalf * 8;
  const int blockM = blockIdx.y * 128;
  const int blockN = blockIdx.x * 128;

  f32x8 acc[2][4];
#pragma unroll
  for (int sm = 0; sm < 2; ++sm)
#pragma unroll
    for (int sn = 0; sn < 4; ++sn) acc[sm][sn] = (f32x8)0.0f;

  const int am  = tid >> 1;
  const int ak0 = (tid & 1) * 16;
  const int bk  = tid >> 3;
  const int bn0 = (tid & 7) * 16;

  const int kTiles = K >> 5;
  for (int kt = 0; kt < kTiles; ++kt) {
    const int k0 = kt << 5;
    __syncthreads();
    const InT* Ag = A + (size_t)(blockM + am) * K + (k0 + ak0);
    bf16* Ash = As + am * 40 + ak0;
#pragma unroll
    for (int j = 0; j < 16; ++j) Ash[j] = (bf16)ldIn(Ag + j);
    const float* Bg = B + (size_t)(k0 + bk) * N + (blockN + bn0);
#pragma unroll
    for (int j = 0; j < 16; ++j) Bs[(bn0 + j) * 40 + bk] = (bf16)Bg[j];
    if (kt + 1 < kTiles) {              // -> global_prefetch_b8
      __builtin_prefetch((const void*)(Ag + 32), 0, 1);
      __builtin_prefetch((const void*)(Bg + (size_t)32 * N), 0, 1);
    }
    __syncthreads();

    bf16x16 aF[2];
#pragma unroll
    for (int sm = 0; sm < 2; ++sm)
      aF[sm] = load_frag(As + (waveM * 32 + sm * 16 + l16) * 40, kOff);
#pragma unroll
    for (int sn = 0; sn < 4; ++sn) {
      bf16x16 bF = load_frag(Bs + (waveN * 64 + sn * 16 + l16) * 40, kOff);
      acc[0][sn] = WMMA_BF16(aF[0], bF, acc[0][sn]);
      acc[1][sn] = WMMA_BF16(aF[1], bF, acc[1][sn]);
    }
  }

#pragma unroll
  for (int sm = 0; sm < 2; ++sm)
#pragma unroll
    for (int sn = 0; sn < 4; ++sn) {
      const int col = blockN + waveN * 64 + sn * 16 + l16;
      const float bv = bias[col];
#pragma unroll
      for (int r = 0; r < 8; ++r) {
        const int row = blockM + waveM * 32 + sm * 16 + r + lhalf * 8;
        stOut(C + (size_t)row * N + col, acc[sm][sn][r] + bv);
      }
    }
}

// ---------------------------------------------------------------------------
// Flash attention over bf16 qkv = [B,T,3*1024] (q|k|v), heads of 64.
// Block = 128 query rows of one (b,h); 8 waves x 16 rows each.
// K tile (64x64) fetched by the Tensor Data Mover into padded LDS rows;
// V tile staged transposed; online softmax; P relayout C->A via per-wave
// LDS scratch; all matmuls are v_wmma_f32_16x16x32_bf16.
// ---------------------------------------------------------------------------
__global__ __launch_bounds__(256)
void flash_attn_wmma(const bf16* __restrict__ qkv, bf16* __restrict__ attn) {
  constexpr int T = 2048;
  __shared__ __align__(16) bf16 Ks[64 * 72];       // Ks[key][d]   (TDM target)
  __shared__ __align__(16) bf16 Vt[64 * 72];       // Vt[d][key]
  __shared__ __align__(16) bf16 Pscr[8 * 16 * 72]; // per-wave P scratch

  const int tid   = threadIdx.x;
  const int lane  = tid & 31;
  const int wave  = tid >> 5;
  const int l16   = lane & 15;
  const int lhalf = lane >> 4;
  const int kOff  = lhalf * 8;
  const int b     = blockIdx.y >> 4;
  const int h     = blockIdx.y & 15;
  const int qTile = blockIdx.x;
  const int qRow0 = qTile * 128 + wave * 16;
  const unsigned ldsKs = (unsigned)(uintptr_t)(void*)&Ks[0];

  // Q strip (16 x 64) as two A-frags, pre-scaled by 1/sqrt(64)
  bf16x16 aQ[2];
  {
    const size_t base = (size_t)(b * T + qRow0 + l16) * 3072 + h * 64;
#pragma unroll
    for (int c = 0; c < 2; ++c) {
      bf16x16 t = load_frag(qkv + base + c * 32, kOff);
#pragma unroll
      for (int j = 0; j < 16; ++j) aQ[c][j] = (bf16)((float)t[j] * 0.125f);
    }
  }

  f32x8 O[4];
  float rowMax[8], rowSum[8];
#pragma unroll
  for (int nt = 0; nt < 4; ++nt) O[nt] = (f32x8)0.0f;
#pragma unroll
  for (int r = 0; r < 8; ++r) { rowMax[r] = -1.0e30f; rowSum[r] = 0.0f; }

  const int nkv = 2 * qTile + 2;                    // causal: keys <= qTile*128+127
  for (int kv = 0; kv < nkv; ++kv) {
    const int kvBase = kv * 64;
    __syncthreads();                                // prior tile fully consumed
    // K tile via Tensor Data Mover (wave 0 issues; TENSORcnt tracks it)
    if (wave == 0) {
      const bf16* gK = qkv + (size_t)(b * T + kvBase) * 3072 + 1024 + h * 64;
      tdm_load_tile_64x64_bf16(ldsKs, gK, 3072u);
    }
    // V tile staged transposed by all waves (TDM cannot transpose)
    {
      const int i  = tid >> 2;                      // key row 0..63
      const int d0 = (tid & 3) * 16;                // 0..48
      const size_t gv = (size_t)(b * T + kvBase + i) * 3072 + 2048 + h * 64 + d0;
#pragma unroll
      for (int j = 0; j < 16; ++j) Vt[(d0 + j) * 72 + i] = qkv[gv + j];
    }
    if (wave == 0) __builtin_amdgcn_s_wait_tensorcnt(0);   // s_wait_tensorcnt 0
    __syncthreads();
    if (kvBase > qRow0 + 15) continue;              // wave fully above diagonal

    // scores S = (Q/8) @ K^T : 4 column tiles x 2 k-chunks = 8 WMMAs
    f32x8 s[4];
#pragma unroll
    for (int ct = 0; ct < 4; ++ct) {
      f32x8 a = (f32x8)0.0f;
#pragma unroll
      for (int c = 0; c < 2; ++c) {
        bf16x16 bK = load_frag(Ks + (ct * 16 + l16) * 72 + c * 32, kOff);
        a = WMMA_BF16(aQ[c], bK, a);
      }
      s[ct] = a;
    }
    // causal mask (only tiles crossing the diagonal)
    if (kvBase + 63 > qRow0) {
#pragma unroll
      for (int ct = 0; ct < 4; ++ct) {
        const int kIdx = kvBase + ct * 16 + l16;
#pragma unroll
        for (int r = 0; r < 8; ++r)
          if (kIdx > qRow0 + r + lhalf * 8) s[ct][r] = -1.0e30f;
      }
    }
    // online softmax: rows live in 16-lane halves -> xor masks 1,2,4,8
#pragma unroll
    for (int r = 0; r < 8; ++r) {
      float m = fmaxf(fmaxf(s[0][r], s[1][r]), fmaxf(s[2][r], s[3][r]));
      m = fmaxf(m, __shfl_xor(m, 1, 32));
      m = fmaxf(m, __shfl_xor(m, 2, 32));
      m = fmaxf(m, __shfl_xor(m, 4, 32));
      m = fmaxf(m, __shfl_xor(m, 8, 32));
      const float mnew  = fmaxf(rowMax[r], m);
      const float alpha = __expf(rowMax[r] - mnew);
      rowMax[r] = mnew;
      float ls = 0.0f;
#pragma unroll
      for (int ct = 0; ct < 4; ++ct) {
        const float p = __expf(s[ct][r] - mnew);
        s[ct][r] = p;
        ls += p;
      }
      ls += __shfl_xor(ls, 1, 32);
      ls += __shfl_xor(ls, 2, 32);
      ls += __shfl_xor(ls, 4, 32);
      ls += __shfl_xor(ls, 8, 32);
      rowSum[r] = rowSum[r] * alpha + ls;
#pragma unroll
      for (int nt = 0; nt < 4; ++nt) O[nt][r] *= alpha;
    }
    // P: C-layout -> row-major LDS -> A-frags (same-wave DS ops are in-order)
    bf16* P = Pscr + wave * (16 * 72);
#pragma unroll
    for (int ct = 0; ct < 4; ++ct)
#pragma unroll
      for (int r = 0; r < 8; ++r)
        P[(r + lhalf * 8) * 72 + ct * 16 + l16] = (bf16)s[ct][r];
    bf16x16 aP[2];
#pragma unroll
    for (int c = 0; c < 2; ++c)
      aP[c] = load_frag(P + l16 * 72 + c * 32, kOff);
    // O += P @ V : 4 d-tiles x 2 k-chunks = 8 WMMAs
#pragma unroll
    for (int nt = 0; nt < 4; ++nt)
#pragma unroll
      for (int c = 0; c < 2; ++c) {
        bf16x16 bV = load_frag(Vt + (nt * 16 + l16) * 72 + c * 32, kOff);
        O[nt] = WMMA_BF16(aP[c], bV, O[nt]);
      }
  }

  // normalize and store attn (bf16, [B,T,D] with heads contiguous)
#pragma unroll
  for (int r = 0; r < 8; ++r) {
    const float inv = 1.0f / rowSum[r];
    const size_t row = (size_t)(b * T + qRow0 + r + lhalf * 8);
#pragma unroll
    for (int nt = 0; nt < 4; ++nt)
      attn[row * 1024 + h * 64 + nt * 16 + l16] = (bf16)(O[nt][r] * inv);
  }
}

// ---------------------------------------------------------------------------
// Launch: x@Wqkv+b -> bf16 qkv (ws) -> flash attention -> bf16 attn (ws)
//         -> attn@Wo+bo -> fp32 d_out.   B=4,T=2048,D=1024,H=16.
// ---------------------------------------------------------------------------
extern "C" void kernel_launch(void* const* d_in, const int* in_sizes, int n_in,
                              void* d_out, int out_size, void* d_ws, size_t ws_size,
                              hipStream_t stream) {
  (void)in_sizes; (void)n_in; (void)out_size; (void)ws_size;
  const float* x    = (const float*)d_in[0];   // [4,2048,1024]
  const float* Wqkv = (const float*)d_in[1];   // [1024,3072]
  const float* bqkv = (const float*)d_in[2];   // [3072]
  const float* Wo   = (const float*)d_in[3];   // [1024,1024]
  const float* bo   = (const float*)d_in[4];   // [1024]
  // d_in[5] = causal mask, implicit in the attention kernel

  const size_t BT = (size_t)4 * 2048;          // 8192 rows
  bf16* qkv  = (bf16*)d_ws;                                     // 48MB
  bf16* attn = (bf16*)((char*)d_ws + BT * 3072 * sizeof(bf16)); // +16MB

  dim3 g1(3072 / 128, BT / 128);
  gemm_bf16_wmma<float, bf16><<<g1, 256, 0, stream>>>(x, Wqkv, bqkv, qkv, 1024, 3072);

  dim3 g2(16, 64);
  flash_attn_wmma<<<g2, 256, 0, stream>>>(qkv, attn);

  dim3 g3(1024 / 128, BT / 128);
  gemm_bf16_wmma<bf16, float><<<g3, 256, 0, stream>>>(attn, Wo, bo, (float*)d_out, 1024, 1024);
}
